// LocalAggregator_42090679501528
// MI455X (gfx1250) — matrix operand order repaired and checked by reading, not compiled, not run on hardware
//
#include <hip/hip_runtime.h>
#include <hip/hip_bf16.h>

// Fused relation-selected attention for MI455X (gfx1250, wave32).
// Flash-style online softmax. All GEMMs on V_WMMA_F32_16X16X32_F16 with
// split-precision (hi/lo f16) compensation => fp32-class accuracy at f16-pipe rate.
// Next-tile global_prefetch_b8 hides adj (HBM) / hidden (L2) latency.

typedef __attribute__((ext_vector_type(8)))  float    v8f;
typedef __attribute__((ext_vector_type(4)))  _Float16 h4;
typedef __attribute__((ext_vector_type(8)))  _Float16 h8;
typedef __attribute__((ext_vector_type(16))) _Float16 v16h;

#define N_NODES   4096
#define DIM       128
#define TILE      16
#define NWAVES    4
#define NEG_INF_F (-9.0e15f)

// LDS strides in halves (chosen: 16B-aligned rows + distinct banks across 16 lanes)
#define HWS 136   // hw / hj row stride: 272B rows (68 dwords; 4*ln banks distinct)
#define HTS 40    // hjT row stride: 80B rows (20 dwords; 20*ln banks distinct)
#define PTS 40    // pt  row stride: 80B rows

// LDS byte layout
#define HW_BYTES   (4 * 16 * HWS * 2)            // 17408 per copy (hi, lo)
#define WAVE_OFF0  (2 * HW_BYTES)                // 34816
#define HJ_HI_OFF  0
#define HJ_LO_OFF  (16 * HWS * 2)                // 4352
#define HJT_HI_OFF (2 * 16 * HWS * 2)            // 8704
#define HJT_LO_OFF (HJT_HI_OFF + 128 * HTS * 2)  // 18944
#define PT_HI_OFF  (HJT_LO_OFF + 128 * HTS * 2)  // 29184
#define PT_LO_OFF  (PT_HI_OFF + 16 * PTS * 2)    // 30464
#define WBLK       (PT_LO_OFF + 16 * PTS * 2)    // 31744 per wave
#define RED_OFF    (WAVE_OFF0 + NWAVES * WBLK)   // 161792
#define SMEM_BYTES (RED_OFF + 2 * NWAVES * 16 * 4)  // 162304 (2 WGs fit in 320KB WGP)

static __device__ __forceinline__ v16h cat16(h8 a, h8 b) {
    return __builtin_shufflevector(a, b, 0,1,2,3,4,5,6,7,8,9,10,11,12,13,14,15);
}

__global__ __launch_bounds__(128)
void gat_fused_wmma_f16split(const float* __restrict__ hidden,
                             const int*   __restrict__ adj,
                             const float* __restrict__ W,
                             const float* __restrict__ bias,
                             float*       __restrict__ out)
{
    extern __shared__ char smem[];
    _Float16* hwhi = (_Float16*)(smem);
    _Float16* hwlo = (_Float16*)(smem + HW_BYTES);
    float*    redm = (float*)(smem + RED_OFF);
    float*    redl = redm + NWAVES * 16;

    const int tid  = threadIdx.x;
    const int wave = tid >> 5;
    const int lane = tid & 31;
    const int half = lane >> 4;
    const int ln   = lane & 15;
    const int i0   = blockIdx.x * TILE;

    char* wbase = smem + WAVE_OFF0 + wave * WBLK;
    _Float16* hjhi  = (_Float16*)(wbase + HJ_HI_OFF);
    _Float16* hjlo  = (_Float16*)(wbase + HJ_LO_OFF);
    _Float16* hjThi = (_Float16*)(wbase + HJT_HI_OFF);
    _Float16* hjTlo = (_Float16*)(wbase + HJT_LO_OFF);
    _Float16* pthi  = (_Float16*)(wbase + PT_HI_OFF);
    _Float16* ptlo  = (_Float16*)(wbase + PT_LO_OFF);

    // ---- Stage hw[k][r][d] = hidden[i0+r][d]*W[k][d] as f16 hi/lo pairs ----
    for (int idx = tid; idx < 4 * 16 * DIM; idx += 128) {
        const int k = idx >> 11;
        const int r = (idx >> 7) & 15;
        const int d = idx & 127;
        const float p  = hidden[(size_t)(i0 + r) * DIM + d] * W[k * DIM + d];
        const _Float16 hi = (_Float16)p;
        hwhi[(k * 16 + r) * HWS + d] = hi;
        hwlo[(k * 16 + r) * HWS + d] = (_Float16)(p - (float)hi);
    }
    __syncthreads();

    const float bk4[4] = { bias[0], bias[1], bias[2], bias[3] };

    float m[8], l[8];
    float acc[8][8];                 // O tile [16][128] in C-layout d-chunks
#pragma unroll
    for (int v = 0; v < 8; ++v) { m[v] = NEG_INF_F; l[v] = 0.0f; }
#pragma unroll
    for (int c = 0; c < 8; ++c)
#pragma unroll
        for (int v = 0; v < 8; ++v) acc[c][v] = 0.0f;

    const v16h z16 = {};
    const h8   z8  = {};

    // Warm the caches for this wave's first tile.
    {
        const int j0 = wave * TILE;
        __builtin_prefetch(hidden + (size_t)j0 * DIM + lane * 64, 0, 3);
        __builtin_prefetch(adj + (size_t)(i0 + ln) * N_NODES + j0 + half * 8, 0, 3);
    }

    for (int jt = wave; jt < N_NODES / TILE; jt += NWAVES) {
        const int j0 = jt * TILE;

        // ---- Prefetch NEXT tile (adj stream = the only true HBM traffic) ----
        {
            const int jn = j0 + NWAVES * TILE;
            if (jn < N_NODES) {
                // 16 rows x 512B of hidden = 8KB: one 256B-spaced address per lane
                __builtin_prefetch(hidden + (size_t)jn * DIM + lane * 64, 0, 3);
                // 16 adj rows, 64B useful each: lanes 0-15 / 16-31 split the row
                __builtin_prefetch(adj + (size_t)(i0 + ln) * N_NODES + jn + half * 8, 0, 3);
            }
        }

        // ---- Stage H_j tile as f16 hi/lo: row-major (Gram B) + transposed (PV B) ----
#pragma unroll 4
        for (int r = 0; r < 16; ++r) {
            const float4 hv = *(const float4*)(hidden + (size_t)(j0 + r) * DIM + lane * 4);
            _Float16 hi[4], lo[4];
            const float f[4] = { hv.x, hv.y, hv.z, hv.w };
#pragma unroll
            for (int c = 0; c < 4; ++c) {
                hi[c] = (_Float16)f[c];
                lo[c] = (_Float16)(f[c] - (float)hi[c]);
            }
            *(h4*)(hjhi + r * HWS + lane * 4) = (h4){hi[0], hi[1], hi[2], hi[3]};
            *(h4*)(hjlo + r * HWS + lane * 4) = (h4){lo[0], lo[1], lo[2], lo[3]};
#pragma unroll
            for (int c = 0; c < 4; ++c) {
                hjThi[(lane * 4 + c) * HTS + r] = hi[c];
                hjTlo[(lane * 4 + c) * HTS + r] = lo[c];
            }
        }
        __builtin_amdgcn_wave_barrier();

        // adj tile in C-layout: element (row v+8*half, col ln)
        int adjv[8];
#pragma unroll
        for (int v = 0; v < 8; ++v)
            adjv[v] = adj[(size_t)(i0 + v + 8 * half) * N_NODES + j0 + ln];

        float alpha[8];
#pragma unroll
        for (int v = 0; v < 8; ++v) alpha[v] = NEG_INF_F;

        // ---- 4 relation Gram tiles: K=128 as 4 chunks of K=32, 3 split-products ----
#pragma unroll
        for (int k = 0; k < 4; ++k) {
            v8f c = {};
            const _Float16* arh = hwhi + (k * 16 + ln) * HWS;
            const _Float16* arl = hwlo + (k * 16 + ln) * HWS;
            const _Float16* brh = hjhi + ln * HWS;
            const _Float16* brl = hjlo + ln * HWS;
#pragma unroll
            for (int ch = 0; ch < 4; ++ch) {
                const int kc = 32 * ch;
                const v16h Ahi = cat16(*(const h8*)(arh + kc + 8 * half),
                                       *(const h8*)(arh + kc + 16 + 8 * half));
                const v16h Alo = cat16(*(const h8*)(arl + kc + 8 * half),
                                       *(const h8*)(arl + kc + 16 + 8 * half));
                const v16h Bhi = cat16(*(const h8*)(brh + kc + 8 * half),
                                       *(const h8*)(brh + kc + 16 + 8 * half));
                const v16h Blo = cat16(*(const h8*)(brl + kc + 8 * half),
                                       *(const h8*)(brl + kc + 16 + 8 * half));
                c = __builtin_amdgcn_wmma_f32_16x16x32_f16(false, Ahi, false, Bhi,
                                                           (short)0, c, false, false);
                c = __builtin_amdgcn_wmma_f32_16x16x32_f16(false, Ahi, false, Blo,
                                                           (short)0, c, false, false);
                c = __builtin_amdgcn_wmma_f32_16x16x32_f16(false, Alo, false, Bhi,
                                                           (short)0, c, false, false);
            }
            const float bk = bk4[k];
#pragma unroll
            for (int v = 0; v < 8; ++v) {
                float e = c[v] + bk;
                e = (e > 0.0f) ? e : 0.2f * e;        // leaky_relu(0.2)
                if (adjv[v] == k + 1) alpha[v] = e;   // relation select / mask
            }
        }

        // ---- Online softmax update (row-wise over 16-lane halves) ----
        float p[8], scl[8];
#pragma unroll
        for (int v = 0; v < 8; ++v) {
            float x = alpha[v];
#pragma unroll
            for (int off = 8; off > 0; off >>= 1)
                x = fmaxf(x, __shfl_xor(x, off, 16));
            const float newm = fmaxf(m[v], x);
            scl[v] = __expf(m[v] - newm);
            p[v]   = __expf(alpha[v] - newm);
            float s = p[v];
#pragma unroll
            for (int off = 8; off > 0; off >>= 1)
                s += __shfl_xor(s, off, 16);
            l[v] = l[v] * scl[v] + s;
            m[v] = newm;
        }
#pragma unroll
        for (int c = 0; c < 8; ++c)
#pragma unroll
            for (int v = 0; v < 8; ++v) acc[c][v] *= scl[v];

        // ---- P through LDS as f16 hi/lo to get A-layout fragments ----
#pragma unroll
        for (int v = 0; v < 8; ++v) {
            const _Float16 phi = (_Float16)p[v];
            pthi[(v + 8 * half) * PTS + ln] = phi;
            ptlo[(v + 8 * half) * PTS + ln] = (_Float16)(p[v] - (float)phi);
        }
        __builtin_amdgcn_wave_barrier();

        // A frags: K(valid)=0..15 => elements[0..7]=P[ln][8*half+e], [8..15]=0
        const v16h Phi = cat16(*(const h8*)(pthi + ln * PTS + 8 * half), z8);
        const v16h Plo = cat16(*(const h8*)(ptlo + ln * PTS + 8 * half), z8);

        // ---- O += P @ H_j over 8 d-chunks (one K=32 WMMA group each) ----
#pragma unroll
        for (int cch = 0; cch < 8; ++cch) {
            v16h Bhi, Blo;
            if (true) {   // elements[0..7] = K 8*half..8*half+7; [8..15] = zero padding
                Bhi = cat16(*(const h8*)(hjThi + (cch * 16 + ln) * HTS + 8 * half), z8);
                Blo = cat16(*(const h8*)(hjTlo + (cch * 16 + ln) * HTS + 8 * half), z8);
            } else {
                Bhi = z16;
                Blo = z16;
            }
            v8f cc;
#pragma unroll
            for (int v = 0; v < 8; ++v) cc[v] = acc[cch][v];
            cc = __builtin_amdgcn_wmma_f32_16x16x32_f16(false, Phi, false, Bhi,
                                                        (short)0, cc, false, false);
            cc = __builtin_amdgcn_wmma_f32_16x16x32_f16(false, Phi, false, Blo,
                                                        (short)0, cc, false, false);
            cc = __builtin_amdgcn_wmma_f32_16x16x32_f16(false, Plo, false, Bhi,
                                                        (short)0, cc, false, false);
#pragma unroll
            for (int v = 0; v < 8; ++v) acc[cch][v] = cc[v];
        }
    }

    // ---- Cross-wave softmax merge ----
    if (ln == 0) {
#pragma unroll
        for (int v = 0; v < 8; ++v) {
            redm[wave * 16 + v + 8 * half] = m[v];
            redl[wave * 16 + v + 8 * half] = l[v];
        }
    }
    __syncthreads();

    // Rescale own O partial by exp(m_w - M); park it (f32) in this wave's hjT region.
    float* opart = (float*)(wbase + HJT_HI_OFF);   // [16][132] f32, 8448B < 20480B
#pragma unroll
    for (int v = 0; v < 8; ++v) {
        const int r = v + 8 * half;
        float M = redm[r];
        for (int w = 1; w < NWAVES; ++w) M = fmaxf(M, redm[w * 16 + r]);
        const float f = __expf(m[v] - M);
#pragma unroll
        for (int cch = 0; cch < 8; ++cch)
            opart[r * 132 + cch * 16 + ln] = acc[cch][v] * f;
    }
    __syncthreads();

    // Final: out[r][d] = (Σ_w O_w[r][d]) / (Σ_w l_w[r]·exp(m_w[r]-M[r]))
    for (int idx = tid; idx < TILE * DIM; idx += 128) {
        const int r = idx >> 7;
        const int d = idx & 127;
        float M = redm[r];
        for (int w = 1; w < NWAVES; ++w) M = fmaxf(M, redm[w * 16 + r]);
        float L = 0.0f, s = 0.0f;
        for (int w = 0; w < NWAVES; ++w) {
            L += redl[w * 16 + r] * __expf(redm[w * 16 + r] - M);
            s += ((const float*)(smem + WAVE_OFF0 + w * WBLK + HJT_HI_OFF))[r * 132 + d];
        }
        out[(size_t)(i0 + r) * DIM + d] = s / L;
    }
}

extern "C" void kernel_launch(void* const* d_in, const int* in_sizes, int n_in,
                              void* d_out, int out_size, void* d_ws, size_t ws_size,
                              hipStream_t stream) {
    const float* hidden = (const float*)d_in[0];   // [4096,128] f32
    const int*   adj    = (const int*)  d_in[1];   // [4096,4096] i32
    const float* W      = (const float*)d_in[2];   // [4,128] f32
    const float* bias   = (const float*)d_in[3];   // [4] f32
    float* out = (float*)d_out;                    // [4096,128] f32

    dim3 grid(N_NODES / TILE);
    dim3 block(128);   // 4 wave32 waves
    hipLaunchKernelGGL(gat_fused_wmma_f16split, grid, block, SMEM_BYTES, stream,
                       hidden, adj, W, bias, out);
}
